// LinearAttention_71691594105312
// MI455X (gfx1250) — matrix operand order repaired and checked by reading, not compiled, not run on hardware
//
#include <hip/hip_runtime.h>
#include <hip/hip_bf16.h>

// ---------------------------------------------------------------------------
// Linear attention for MI455X (gfx1250), wave32, WMMA f16 path.
// D_MODEL=256, H=8, D=32, B=32, L=4096  -> tokens M = 131072
// ---------------------------------------------------------------------------

typedef _Float16 f16;
typedef __attribute__((ext_vector_type(16))) _Float16 v16h;
typedef __attribute__((ext_vector_type(8)))  _Float16 v8h;
typedef __attribute__((ext_vector_type(8)))  float    v8f;
typedef __attribute__((ext_vector_type(4)))  float    v4f;

#define GAS __attribute__((address_space(1)))

// Explicit global-address-space accessors: force global_load_*/global_store_*
// instead of flat_* (flat ticks both LOADcnt and DScnt and ties up the LDS path).
__device__ __forceinline__ v8h gload_v8h(const f16* p) {
    return *(const GAS v8h*)(unsigned long long)p;
}
__device__ __forceinline__ v4f gload_f4(const float* p) {
    return *(const GAS v4f*)(unsigned long long)p;
}
__device__ __forceinline__ float gload_f(const float* p) {
    return *(const GAS float*)(unsigned long long)p;
}
__device__ __forceinline__ void gstore_h(f16* p, f16 v) {
    *(GAS f16*)(unsigned long long)p = v;
}
__device__ __forceinline__ void gstore_f(float* p, float v) {
    *(GAS float*)(unsigned long long)p = v;
}

#define WMMA_F16(A, B, C) \
    __builtin_amdgcn_wmma_f32_16x16x32_f16(false, (A), false, (B), (short)0, (C), false, false)

// A fragment (16x32 f16): lane holds row M=lane&15; elements 0..7 = K at p0,
// elements 8..15 = K at p0+16.  Both 16B-aligned global loads.
__device__ __forceinline__ v16h load_a16(const f16* p0) {
    v8h x0 = gload_v8h(p0);
    v8h x1 = gload_v8h(p0 + 16);
    v16h r;
#pragma unroll
    for (int i = 0; i < 8; ++i) { r[i] = x0[i]; r[i + 8] = x1[i]; }
    return r;
}

// B fragment (32x16 f16): lane holds col N=lane&15 (+tile); elements 0..7 = K at p,
// elements 8..15 = K at p+16.  Both 16B-aligned global loads.
__device__ __forceinline__ v16h load_b16(const f16* p) {
    v8h x0 = gload_v8h(p);
    v8h x1 = gload_v8h(p + 16);
    v16h r;
#pragma unroll
    for (int i = 0; i < 8; ++i) { r[i] = x0[i]; r[i + 8] = x1[i]; }
    return r;
}

// Branchless elu(x)+1: exp(x) for x<0, x+1 for x>=0 (no exec-mask diamonds).
__device__ __forceinline__ float feature_map(float v) {
    float e = __expf(fminf(v, 0.0f));
    return (v >= 0.0f) ? (v + 1.0f) : e;
}

// ---------------------------------------------------------------------------
// Kernel 0: convert + transpose the four 256x256 f32 weights to f16 wT[n][k].
// grid (256, 4), block 256.
// ---------------------------------------------------------------------------
__global__ void la_convw(const float* __restrict__ w0, const float* __restrict__ w1,
                         const float* __restrict__ w2, const float* __restrict__ w3,
                         f16* t0, f16* t1, f16* t2, f16* t3) {
    const float* src = (blockIdx.y == 0) ? w0 : (blockIdx.y == 1) ? w1
                     : (blockIdx.y == 2) ? w2 : w3;
    f16* dst = (blockIdx.y == 0) ? t0 : (blockIdx.y == 1) ? t1
             : (blockIdx.y == 2) ? t2 : t3;
    int n = blockIdx.x;
    int k = threadIdx.x;
    gstore_h(dst + n * 256 + k, (f16)gload_f(src + k * 256 + n));
}

// ---------------------------------------------------------------------------
// Kernel 1: Q/K/V GEMMs.  1024 blocks x 256 threads (8 waves); each wave owns
// a 16-row M tile, 8 hoisted A fragments, 3 weights x 16 N-tiles x 8 K-steps
// of v_wmma_f32_16x16x32_f16.  Feature map elu(x)+1 applied to Q,K.
// Q stored [tok][256]; K,V stored transposed [b,h,d|e][l] f16.
// ---------------------------------------------------------------------------
__global__ __launch_bounds__(256) void la_qkv(
    const float* __restrict__ x,
    const f16* __restrict__ wqT, const f16* __restrict__ wkT, const f16* __restrict__ wvT,
    const float* __restrict__ bq, const float* __restrict__ bk, const float* __restrict__ bv,
    f16* __restrict__ qf, f16* __restrict__ kt, f16* __restrict__ vt) {
    const int wave = threadIdx.x >> 5;
    const int lane = threadIdx.x & 31;
    const int nl   = lane & 15;
    const int hi   = lane >> 4;
    const int m0   = (blockIdx.x * 8 + wave) * 16;
    const int row  = m0 + nl;                 // global token for A rows

    // Hoist + convert A fragments: x[row, 0..255] f32 -> f16.
    v16h a[8];
    const float* xr = x + (size_t)row * 256;
#pragma unroll
    for (int kc = 0; kc < 8; ++kc) {
        const float* p0 = xr + kc * 32 + hi * 8;
        const float* p1 = p0 + 16;
        v4f f0 = gload_f4(p0);
        v4f f1 = gload_f4(p0 + 4);
        v4f f2 = gload_f4(p1);
        v4f f3 = gload_f4(p1 + 4);
        v16h t;
#pragma unroll
        for (int i = 0; i < 4; ++i) {
            t[i]      = (f16)f0[i];
            t[i + 4]  = (f16)f1[i];
            t[i + 8]  = (f16)f2[i];
            t[i + 12] = (f16)f3[i];
        }
        a[kc] = t;
    }

    const f16*   wTs[3] = {wqT, wkT, wvT};
    const float* bss[3] = {bq, bk, bv};
    const int bb = row >> 12;   // batch (rows in a tile never cross a batch: 4096%16==0)

#pragma unroll
    for (int wi = 0; wi < 3; ++wi) {
        const f16*   wT   = wTs[wi];
        const float* bias = bss[wi];
#pragma unroll 1
        for (int nt = 0; nt < 16; ++nt) {
            const int ncol   = nt * 16 + nl;        // output column this lane owns
            const int kstart = hi * 8;
            const f16* wp = wT + (size_t)ncol * 256 + kstart;
            if (nt < 15)  // prefetch next tile's B column stream (global_prefetch_b8)
                __builtin_prefetch(wp + 16 * 256, 0, 0);
            v8f acc = {};
#pragma unroll
            for (int kc = 0; kc < 8; ++kc) {
                v16h bf = load_b16(wp + kc * 32);
                acc = WMMA_F16(a[kc], bf, acc);
            }
            const float bn = gload_f(bias + ncol);
            const int h = ncol >> 5, d = ncol & 31;
#pragma unroll
            for (int r = 0; r < 8; ++r) {
                const int m   = r + (hi << 3);
                const int tok = m0 + m;
                float v = acc[r] + bn;
                if (wi < 2) v = feature_map(v);
                const f16 hv = (f16)v;
                if (wi == 0) {
                    gstore_h(qf + (size_t)tok * 256 + ncol, hv);
                } else {
                    const int ll = tok & 4095;
                    const size_t idx = (((size_t)bb * 8 + h) * 32 + d) * 4096 + ll;
                    if (wi == 1) gstore_h(kt + idx, hv);
                    else         gstore_h(vt + idx, hv);
                }
            }
        }
    }
}

// ---------------------------------------------------------------------------
// Kernel 2: per-(b,h) kv = K^T V (32x32) and k_sum (32).  256 blocks (one per
// (b,h)), 8 waves split L=4096; 2x2 WMMA tiles; LDS f32 atomics reduce waves.
// kv stored transposed kvT[e][d] f16 for kernel 3's B fragments.
// ---------------------------------------------------------------------------
__global__ __launch_bounds__(256) void la_kv(
    const f16* __restrict__ kt, const f16* __restrict__ vt,
    f16* __restrict__ kvT, float* __restrict__ ksum) {
    __shared__ float s_kv[1024];
    __shared__ float s_ks[32];
    const int tid = threadIdx.x;
    for (int i = tid; i < 1024; i += 256) s_kv[i] = 0.0f;
    if (tid < 32) s_ks[tid] = 0.0f;
    __syncthreads();

    const int bh   = blockIdx.x;
    const f16* Kb = kt + (size_t)bh * 32 * 4096;
    const f16* Vb = vt + (size_t)bh * 32 * 4096;
    const int wave = tid >> 5;
    const int lane = tid & 31;
    const int nl   = lane & 15;
    const int hi   = lane >> 4;

    v8f acc[2][2] = {};
    float ks0 = 0.0f, ks1 = 0.0f;
#pragma unroll 1
    for (int kc = 0; kc < 16; ++kc) {
        const int lb = wave * 512 + kc * 32;
        v16h a0 = load_a16(Kb + (size_t)nl        * 4096 + lb + hi * 8);
        v16h a1 = load_a16(Kb + (size_t)(nl + 16) * 4096 + lb + hi * 8);
        v16h b0 = load_b16(Vb + (size_t)nl        * 4096 + lb + hi * 8);
        v16h b1 = load_b16(Vb + (size_t)(nl + 16) * 4096 + lb + hi * 8);
        acc[0][0] = WMMA_F16(a0, b0, acc[0][0]);
        acc[0][1] = WMMA_F16(a0, b1, acc[0][1]);
        acc[1][0] = WMMA_F16(a1, b0, acc[1][0]);
        acc[1][1] = WMMA_F16(a1, b1, acc[1][1]);
#pragma unroll
        for (int i = 0; i < 16; ++i) { ks0 += (float)a0[i]; ks1 += (float)a1[i]; }
    }
    atomicAdd(&s_ks[nl],      ks0);
    atomicAdd(&s_ks[nl + 16], ks1);
#pragma unroll
    for (int ti = 0; ti < 2; ++ti)
#pragma unroll
        for (int tj = 0; tj < 2; ++tj)
#pragma unroll
            for (int r = 0; r < 8; ++r) {
                const int d = ti * 16 + r + (hi << 3);
                const int e = tj * 16 + nl;
                atomicAdd(&s_kv[d * 32 + e], acc[ti][tj][r]);
            }
    __syncthreads();

    for (int i = tid; i < 1024; i += 256) {
        const int d = i >> 5, e = i & 31;
        gstore_h(kvT + (size_t)bh * 1024 + e * 32 + d, (f16)s_kv[i]);
    }
    if (tid < 32) gstore_f(ksum + bh * 32 + tid, s_ks[tid]);
}

// ---------------------------------------------------------------------------
// Kernel 3: per 16-token tile per head: Q@kv (two WMMAs), normalize by
// q.k_sum + 1e-6, store attn f16 [tok][256].  8192 blocks x 8 waves.
// ---------------------------------------------------------------------------
__global__ __launch_bounds__(256) void la_attn(
    const f16* __restrict__ qf, const f16* __restrict__ kvT,
    const float* __restrict__ ksum, f16* __restrict__ attn) {
    const int wid  = blockIdx.x * 8 + (threadIdx.x >> 5);  // 0..65535
    const int lane = threadIdx.x & 31;
    const int nl   = lane & 15;
    const int hi   = lane >> 4;
    const int h  = wid & 7;
    const int lt = (wid >> 3) & 255;
    const int b  = wid >> 11;
    const int bh = b * 8 + h;
    const size_t tok0 = (size_t)b * 4096 + (size_t)lt * 16;

    const f16* qrow = qf + (tok0 + nl) * 256 + h * 32;
    v16h aq = load_a16(qrow + hi * 8);
    const f16* kvp = kvT + (size_t)bh * 1024;
    v16h b0 = load_b16(kvp + (size_t)nl * 32 + hi * 8);
    v16h b1 = load_b16(kvp + (size_t)(nl + 16) * 32 + hi * 8);
    v8f acc0 = {}, acc1 = {};
    acc0 = WMMA_F16(aq, b0, acc0);
    acc1 = WMMA_F16(aq, b1, acc1);

    // normalizer for token m = lane&15 (lanes r and r+16 compute the same m,
    // so shfl from src lane in [0,16) is always valid).  Vector loads.
    const float* ksp = ksum + bh * 32;
    float norm = 1e-6f;
#pragma unroll
    for (int c = 0; c < 4; ++c) {
        v8h qv = gload_v8h(qrow + c * 8);
        v4f k0 = gload_f4(ksp + c * 8);
        v4f k1 = gload_f4(ksp + c * 8 + 4);
#pragma unroll
        for (int i = 0; i < 4; ++i)
            norm += (float)qv[i] * k0[i] + (float)qv[i + 4] * k1[i];
    }
    const float inv = 1.0f / norm;

#pragma unroll
    for (int r = 0; r < 8; ++r) {
        const int m = r + (hi << 3);
        const float iv = __shfl(inv, m, 32);
        f16* op = attn + (tok0 + m) * 256 + h * 32 + nl;
        gstore_h(op,      (f16)(acc0[r] * iv));
        gstore_h(op + 16, (f16)(acc1[r] * iv));
    }
}

// ---------------------------------------------------------------------------
// Kernel 4: out = attn @ Wo + bo, f32 output.  Same shape as kernel 1.
// ---------------------------------------------------------------------------
__global__ __launch_bounds__(256) void la_out(
    const f16* __restrict__ attn, const f16* __restrict__ woT,
    const float* __restrict__ bo, float* __restrict__ out) {
    const int wave = threadIdx.x >> 5;
    const int lane = threadIdx.x & 31;
    const int nl   = lane & 15;
    const int hi   = lane >> 4;
    const int m0   = (blockIdx.x * 8 + wave) * 16;
    const int row  = m0 + nl;

    v16h a[8];
    const f16* ar = attn + (size_t)row * 256;
#pragma unroll
    for (int kc = 0; kc < 8; ++kc)
        a[kc] = load_a16(ar + kc * 32 + hi * 8);

#pragma unroll 1
    for (int nt = 0; nt < 16; ++nt) {
        const int ncol   = nt * 16 + nl;
        const int kstart = hi * 8;
        const f16* wp = woT + (size_t)ncol * 256 + kstart;
        if (nt < 15)
            __builtin_prefetch(wp + 16 * 256, 0, 0);
        v8f acc = {};
#pragma unroll
        for (int kc = 0; kc < 8; ++kc) {
            v16h bf = load_b16(wp + kc * 32);
            acc = WMMA_F16(a[kc], bf, acc);
        }
        const float bn = gload_f(bo + ncol);
#pragma unroll
        for (int r = 0; r < 8; ++r) {
            const int m = r + (hi << 3);
            gstore_f(out + (size_t)(m0 + m) * 256 + ncol, acc[r] + bn);
        }
    }
}

// ---------------------------------------------------------------------------
// Host-side launch.
// ---------------------------------------------------------------------------
extern "C" void kernel_launch(void* const* d_in, const int* in_sizes, int n_in,
                              void* d_out, int out_size, void* d_ws, size_t ws_size,
                              hipStream_t stream) {
    (void)in_sizes; (void)n_in; (void)out_size; (void)ws_size;

    const float* x  = (const float*)d_in[0];
    const float* wq = (const float*)d_in[1];
    const float* bq = (const float*)d_in[2];
    const float* wk = (const float*)d_in[3];
    const float* bk = (const float*)d_in[4];
    const float* wv = (const float*)d_in[5];
    const float* bv = (const float*)d_in[6];
    const float* wo = (const float*)d_in[7];
    const float* bo = (const float*)d_in[8];
    float* out = (float*)d_out;

    const size_t TOKS = 32ull * 4096ull;      // 131072
    const size_t BIG  = TOKS * 256ull;        // 33,554,432 halfs = 64 MB

    f16* qf  = (f16*)d_ws;                    // Q f16 [tok][256]
    f16* ktf = qf  + BIG;                     // K^T f16 [b,h,d][l]
    f16* vtf = ktf + BIG;                     // V^T f16 [b,h,e][l]
    f16* attn = ktf;                          // reuse K^T region after kernel 2
    f16* wqT = vtf + BIG;                     // 4 x 64K halfs weight transposes
    f16* wkT = wqT + 65536;
    f16* wvT = wkT + 65536;
    f16* woT = wvT + 65536;
    f16* kvT = woT + 65536;                   // 256 * 1024 halfs
    float* ksumf = (float*)(kvT + 262144);    // 256 * 32 floats

    la_convw<<<dim3(256, 4), 256, 0, stream>>>(wq, wk, wv, wo, wqT, wkT, wvT, woT);
    la_qkv  <<<1024, 256, 0, stream>>>(x, wqT, wkT, wvT, bq, bk, bv, qf, ktf, vtf);
    la_kv   <<<256, 256, 0, stream>>>(ktf, vtf, kvT, ksumf);
    la_attn <<<8192, 256, 0, stream>>>(qf, kvT, ksumf, attn);
    la_out  <<<1024, 256, 0, stream>>>(attn, woT, bo, out);
}
